// Network_71451075936980
// MI455X (gfx1250) — compile-verified, matrix operand order checked by reading, not run on hardware
//
#include <hip/hip_runtime.h>
#include <math.h>

// Problem dimensions (from reference)
#define DEPTH      512
#define INPUT_SZ   256
#define STATE_SZ   4096
#define NSCALES    8

// Scan decomposition: N = 256 tiles of 16; K = 128 chunks of 32.
// 128 blocks x 128 threads (4 waves). Block b owns tiles {2b, 2b+1} with the
// weights for both tiles (256 KB bf16) cached in LDS for all 512 steps.
// Wave w computes BOTH tiles over K-quarter [32w, 32w+32) -> each A fragment
// feeds 2 WMMAs (A L2 traffic halved); partial tiles reduced through LDS.
#define SCAN_BLOCKS  128
#define SCAN_THREADS 128
#define KQ_CHUNKS    32

#define SMEM_B_BYTES   (262144u)                 // 2 tiles * 128 KB bf16
#define SMEM_RED_OFF   SMEM_B_BYTES
#define SMEM_BYTES     (SMEM_B_BYTES + 8192u)    // + 4 waves * 2 tiles * 1 KB

// Workspace layout (bytes)
#define OFF_BPACK  ((size_t)0)                       // 32 MB bf16 B-fragments
#define OFF_BIAS   ((size_t)33554432)                // 8 MB fp32 biases
#define OFF_A0     (OFF_BIAS + (size_t)8388608)      // 128 KB A-fragment ping
#define OFF_A1     (OFF_A0 + (size_t)131072)         // 128 KB A-fragment pong
#define OFF_BAR    (OFF_A1 + (size_t)131072)         // barrier counter

typedef __attribute__((ext_vector_type(16))) __bf16  v16bf;
typedef __attribute__((ext_vector_type(8)))  __bf16  v8bf;
typedef __attribute__((ext_vector_type(8)))  float   v8f;
typedef __attribute__((ext_vector_type(4)))  unsigned int v4u;
typedef __attribute__((ext_vector_type(8)))  int     v8i;
typedef __attribute__((ext_vector_type(4)))  int     v4i;

__device__ __forceinline__ unsigned short f2bf(float f) {
    unsigned u = __float_as_uint(f);
    unsigned r = (u + 0x7FFFu + ((u >> 16) & 1u)) >> 16;   // round-to-nearest-even
    return (unsigned short)r;
}

// A-fragment (16x32 bf16, ISA 7.12.2) element position for (row m, K kk):
// lane + element index e (0..15) within the lane's 8 dwords.
__device__ __forceinline__ void frag_pos(unsigned m, unsigned kk,
                                         unsigned& lane, unsigned& e) {
    unsigned hi = ((kk & 15u) >= 8u) ? 16u : 0u;
    unsigned vg = ((kk & 7u) >> 1) + 4u * (kk >> 4);
    lane = m + hi;
    e = vg * 2u + (kk & 1u);
}

// A buffers (global): chunk-contiguous, 32 B per lane: half index within chunk.
__device__ __forceinline__ unsigned afrag_half_idx(unsigned m, unsigned kk) {
    unsigned lane, e; frag_pos(m, kk, lane, e);
    return lane * 16u + e;
}
// B pack (destined for LDS): each 1 KB chunk stored as two 512 B halves with
// 16 B per lane -> conflict-free ds_load_b128 (half-wave covers all 64 banks).
__device__ __forceinline__ unsigned bfrag_half_idx(unsigned n, unsigned kk) {
    unsigned lane, e; frag_pos(n, kk, lane, e);
    return (e >> 3) * 256u + lane * 8u + (e & 7u);
}

// ---------------------------------------------------------------------------
// Pack W_lin (fp32 row-major [i][k]) into bf16 B-fragments in the LDS layout.
// ---------------------------------------------------------------------------
__global__ void pack_wlin_kernel(const float* __restrict__ W,
                                 unsigned short* __restrict__ bpack) {
    unsigned idx = blockIdx.x * 256u + threadIdx.x;         // 0 .. 4096*4096-1
    unsigned k = idx & 4095u;
    unsigned n = idx >> 12;
    unsigned short v = f2bf(W[(size_t)n * STATE_SZ + k]);
    unsigned nt = n >> 4, nn = n & 15u;
    unsigned c  = k >> 5, kk = k & 31u;
    bpack[(size_t)nt * 65536u + (size_t)c * 512u + bfrag_half_idx(nn, kk)] = v;
}

// ---------------------------------------------------------------------------
// biases[t][i] = sum_j W_in[i][j] * seq[t][j]  (1 GFLOP, off critical path)
// ---------------------------------------------------------------------------
__global__ void bias_kernel(const float* __restrict__ W_in,
                            const float* __restrict__ seq,
                            float* __restrict__ bias) {
    __shared__ float sseq[16 * INPUT_SZ];
    const int tid = threadIdx.x;
    const int t0  = blockIdx.y * 16;
    #pragma unroll
    for (int r = 0; r < 16; ++r)
        sseq[r * INPUT_SZ + tid] = seq[(size_t)(t0 + r) * INPUT_SZ + tid];
    __syncthreads();

    const int i = blockIdx.x * 256 + tid;
    const float4* wrow = (const float4*)(W_in + (size_t)i * INPUT_SZ);
    float acc[16];
    #pragma unroll
    for (int r = 0; r < 16; ++r) acc[r] = 0.0f;

    for (int j4 = 0; j4 < INPUT_SZ / 4; ++j4) {
        float4 w = wrow[j4];
        #pragma unroll
        for (int r = 0; r < 16; ++r) {
            const float* s = &sseq[r * INPUT_SZ + j4 * 4];
            acc[r] += w.x * s[0] + w.y * s[1] + w.z * s[2] + w.w * s[3];
        }
    }
    #pragma unroll
    for (int r = 0; r < 16; ++r)
        bias[(size_t)(t0 + r) * STATE_SZ + i] = acc[r];
}

// ---------------------------------------------------------------------------
// Broadcast initial state into A-fragment rows 0..7 of buffer 0 (rows 8..15
// of both buffers stay zero via memset).
// ---------------------------------------------------------------------------
__global__ void init_state_kernel(const float* __restrict__ state,
                                  unsigned short* __restrict__ a0) {
    unsigned k = blockIdx.x * 256u + threadIdx.x;           // 0..4095
    unsigned short v = f2bf(state[k]);
    unsigned c = k >> 5, kk = k & 31u;
    #pragma unroll
    for (unsigned s = 0; s < NSCALES; ++s)
        a0[(size_t)c * 512u + afrag_half_idx(s, kk)] = v;
}

// ---------------------------------------------------------------------------
// Persistent scan. LDS-resident weights (TDM-filled once), A from global
// (L2), 64 x v_wmma per wave per step, LDS K-reduction, erf epilogue,
// grid-wide atomic barrier, ping-pong A-fragment buffers.
// ---------------------------------------------------------------------------
__global__ void __launch_bounds__(SCAN_THREADS, 1)
scan_kernel(const unsigned short* __restrict__ bpack_,
            const float* __restrict__ bias,
            const float* __restrict__ wscale_in,
            unsigned short* __restrict__ a0,
            unsigned short* __restrict__ a1,
            unsigned int* __restrict__ bar,
            float* __restrict__ out) {
    extern __shared__ char smem[];
    const int tid   = threadIdx.x;
    const int lane  = tid & 31;
    const int wave  = tid >> 5;            // 0..3 = K-quarter
    const int nt0   = blockIdx.x * 2;

    // ---- one-time fill: 256 KB of B-fragments -> LDS via Tensor Data Mover
    const unsigned long long gsrc =
        (unsigned long long)(const void*)(bpack_ + (size_t)nt0 * 65536u);
#if defined(__AMDGCN__) && __has_builtin(__builtin_amdgcn_tensor_load_to_lds) && __has_builtin(__builtin_amdgcn_s_wait_tensorcnt)
    if (wave == 0) {
        const unsigned ldsa = (unsigned)(unsigned long long)(void*)smem;
        // D# group0: count=1 | lds_addr | global_addr | type=2
        v4u g0 = { 1u, ldsa, (unsigned)gsrc,
                   (unsigned)(gsrc >> 32) | (2u << 30) };
        // D# group1: data_size=8B; 1-D tensor/tile of 32768 elements (256 KB)
        v8i g1 = { (int)(3u << 16),          // data_size=3 (8B)
                   (int)(0x8000u << 16),     // tensor_dim0=32768 (low 16)
                   (int)(1u << 16),          // tensor_dim0 hi=0, tensor_dim1=1
                   (int)(0x8000u << 16),     // tile_dim0=32768
                   1,                        // tile_dim1=1
                   32768, 0, 0 };            // tensor_dim0_stride
        v4i gz4 = { 0, 0, 0, 0 };
        v8i gz8 = { 0, 0, 0, 0, 0, 0, 0, 0 };
        __builtin_amdgcn_tensor_load_to_lds(g0, g1, gz4, gz4, gz8, 0);
        __builtin_amdgcn_s_wait_tensorcnt(0);
    }
#else
    {
        float4* dst = (float4*)smem;
        const float4* src = (const float4*)(const void*)gsrc;
        for (unsigned idx = tid; idx < SMEM_B_BYTES / 16u; idx += SCAN_THREADS)
            dst[idx] = src[idx];
    }
#endif
    __syncthreads();

    float wsc[NSCALES];
    #pragma unroll
    for (int r = 0; r < NSCALES; ++r) wsc[r] = wscale_in[r];

    // LDS B pointers (16B half-fragments): chunk = 64 v8bf, tile = 128 chunks.
    const unsigned cbase = (unsigned)wave * KQ_CHUNKS;
    const v8bf* bl = (const v8bf*)smem + (size_t)cbase * 64u + lane;
    float* red = (float*)(smem + SMEM_RED_OFF);     // [wave][tile][lane*8]

    unsigned short* abuf0 = a0;
    unsigned short* abuf1 = a1;
    const unsigned nblocks = gridDim.x;

    // Epilogue assignment: wave 0 -> tile0, wave 1 -> tile1.
    const int nte = nt0 + (wave & 1);
    const int ie  = nte * 16 + (lane & 15);
    const unsigned kk  = (unsigned)ie & 31u;
    const unsigned hi  = ((kk & 15u) >= 8u) ? 16u : 0u;
    const unsigned vgh = (((kk & 7u) >> 1) + 4u * (kk >> 4)) * 2u + (kk & 1u);
    const unsigned sbase = ((unsigned)ie >> 5) * 512u + hi * 16u + vgh;

    for (int t = 0; t < DEPTH; ++t) {
        const v16bf* af =
            (const v16bf*)((t & 1) ? abuf1 : abuf0) + (size_t)cbase * 32u + lane;

        v8f acc0 = {}, acc1 = {};        // one chain per N-tile
        #pragma unroll 4
        for (int c = 0; c < KQ_CHUNKS; ++c) {
            v16bf A = af[c * 32];                          // global (L2), 32B/lane
            v8bf b0l = bl[c * 64];                         // tile0 lo half
            v8bf b0h = bl[c * 64 + 32];                    // tile0 hi half
            v8bf b1l = bl[c * 64 + 8192];                  // tile1 (+128KB)
            v8bf b1h = bl[c * 64 + 8192 + 32];
            v16bf B0 = __builtin_shufflevector(b0l, b0h,
                0,1,2,3,4,5,6,7,8,9,10,11,12,13,14,15);
            v16bf B1 = __builtin_shufflevector(b1l, b1h,
                0,1,2,3,4,5,6,7,8,9,10,11,12,13,14,15);
            acc0 = __builtin_amdgcn_wmma_f32_16x16x32_bf16(
                false, A, false, B0, (short)0, acc0, false, false);
            acc1 = __builtin_amdgcn_wmma_f32_16x16x32_bf16(
                false, A, false, B1, (short)0, acc1, false, false);
        }

        // publish partial tiles: red[wave][tile][lane*8 + r]
        {
            float4* p0 = (float4*)&red[((unsigned)wave * 2u + 0u) * 256u + lane * 8u];
            float4* p1 = (float4*)&red[((unsigned)wave * 2u + 1u) * 256u + lane * 8u];
            p0[0] = make_float4(acc0[0], acc0[1], acc0[2], acc0[3]);
            p0[1] = make_float4(acc0[4], acc0[5], acc0[6], acc0[7]);
            p1[0] = make_float4(acc1[0], acc1[1], acc1[2], acc1[3]);
            p1[1] = make_float4(acc1[4], acc1[5], acc1[6], acc1[7]);
        }
        __syncthreads();

        if (wave < 2) {                  // wave w reduces + finishes tile w
            float acc[8];
            #pragma unroll
            for (int r = 0; r < 8; ++r) acc[r] = 0.0f;
            #pragma unroll
            for (int w2 = 0; w2 < 4; ++w2) {
                const float4* p =
                    (const float4*)&red[((unsigned)w2 * 2u + (unsigned)wave) * 256u + lane * 8u];
                float4 r0 = p[0], r1 = p[1];
                acc[0] += r0.x; acc[1] += r0.y; acc[2] += r0.z; acc[3] += r0.w;
                acc[4] += r1.x; acc[5] += r1.y; acc[6] += r1.z; acc[7] += r1.w;
            }
            if (lane < 16) {             // lanes 0-15 hold rows M=0..7 (scales)
                unsigned short* anext = (t & 1) ? abuf0 : abuf1;
                const float bt = bias[(size_t)t * STATE_SZ + ie];
                if (t + 1 < DEPTH)
                    __builtin_prefetch(&bias[(size_t)(t + 1) * STATE_SZ + ie], 0, 0);
                #pragma unroll
                for (int r = 0; r < NSCALES; ++r) {
                    float pre = wsc[r] * acc[r] + bt;
                    float v = erff(pre) * 0.015625f;   // 1/sqrt(4096)
                    out[(size_t)r * (DEPTH * (size_t)STATE_SZ) +
                        (size_t)t * STATE_SZ + ie] = v;
                    anext[sbase + (unsigned)r * 16u] = f2bf(v);
                }
            }
        }

        // grid-wide barrier: all step-t writes visible before step t+1
        __threadfence();
        __syncthreads();
        if (tid == 0) {
            atomicAdd(bar, 1u);
            const unsigned target = nblocks * (unsigned)(t + 1);
            while (__hip_atomic_load(bar, __ATOMIC_ACQUIRE,
                                     __HIP_MEMORY_SCOPE_AGENT) < target) {
                __builtin_amdgcn_s_sleep(1);
            }
        }
        __syncthreads();
        __threadfence();
    }
}

// ---------------------------------------------------------------------------
extern "C" void kernel_launch(void* const* d_in, const int* in_sizes, int n_in,
                              void* d_out, int out_size, void* d_ws, size_t ws_size,
                              hipStream_t stream) {
    const float* seq    = (const float*)d_in[0];   // (512, 256)
    const float* state  = (const float*)d_in[1];   // (4096,)
    const float* W_in   = (const float*)d_in[2];   // (4096, 256)
    const float* W_lin  = (const float*)d_in[3];   // (4096, 4096)
    const float* wscale = (const float*)d_in[4];   // (8,)
    float* out = (float*)d_out;                    // (8, 512, 4096) fp32

    char* ws = (char*)d_ws;
    unsigned short* bpack = (unsigned short*)(ws + OFF_BPACK);
    float*          bias  = (float*)(ws + OFF_BIAS);
    unsigned short* a0    = (unsigned short*)(ws + OFF_A0);
    unsigned short* a1    = (unsigned short*)(ws + OFF_A1);
    unsigned int*   bar   = (unsigned int*)(ws + OFF_BAR);

    // Opt in to >64KB dynamic LDS (WGP has 320 KB). Host-side, capture-safe,
    // deterministic (same call every launch).
    (void)hipFuncSetAttribute((const void*)scan_kernel,
                              hipFuncAttributeMaxDynamicSharedMemorySize,
                              (int)SMEM_BYTES);

    // Zero both A-fragment buffers (rows 8..15 must stay 0) + barrier counter.
    (void)hipMemsetAsync(ws + OFF_A0, 0, 131072u * 2 + 64, stream);

    pack_wlin_kernel<<<65536, 256, 0, stream>>>(W_lin, bpack);
    bias_kernel<<<dim3(16, 32), 256, 0, stream>>>(W_in, seq, bias);
    init_state_kernel<<<16, 256, 0, stream>>>(state, a0);
    scan_kernel<<<SCAN_BLOCKS, SCAN_THREADS, SMEM_BYTES, stream>>>(
        bpack, bias, wscale, a0, a1, bar, out);
}
